// LinearTransformer_90640989814974
// MI455X (gfx1250) — matrix operand order repaired and checked by reading, not compiled, not run on hardware
//
#include <hip/hip_runtime.h>
#include <hip/hip_bf16.h>

// ---------------------------------------------------------------------------
// MI455X (gfx1250) linear-attention transformer block, bf16 WMMA everywhere.
// B=4, N=16384, D=512, H=8, DK=DV=64, HID=2048.
//  - weights pre-transposed/converted to bf16 once (WT[n][k])
//  - GEMM staging uses gfx1250 GLOBAL_LOAD_ASYNC_TO_LDS_B128, double-buffered
//    (ASYNCcnt / s_wait_asynccnt), overlapping copy with WMMA compute
//  - QKV epilogue writes pk/v transposed (packed 16B stores) so the kv
//    reduction runs LDS-free with direct-from-global WMMA fragments
//  - ksum folded into kv via an extra ones-column (vT padded to 80 rows)
// ---------------------------------------------------------------------------

typedef __bf16 bf16;
typedef __attribute__((ext_vector_type(16))) __bf16 v16bf;
typedef __attribute__((ext_vector_type(8)))  __bf16 bf16x8;
typedef __attribute__((ext_vector_type(8)))  float  v8f;

#define MTOK   65536     // B*N tokens
#define DMODEL 512
#define NSEQ   16384
#define HID    2048

__device__ __forceinline__ v16bf make_frag(bf16x8 lo, bf16x8 hi) {
    union { v16bf v; bf16x8 h[2]; } u;
    u.h[0] = lo; u.h[1] = hi;
    return u.v;
}

// A fragment (16x32 bf16): lane l<16: row l, K=0..7 / 16..23 ; lane l+16: K=8..15 / 24..31
__device__ __forceinline__ v16bf a_frag(const bf16* t, int pitch, int kc) {
    const int lane = threadIdx.x & 31;
    const bf16* p = t + (size_t)(lane & 15) * pitch + kc + ((lane >> 4) * 8);
    return make_frag(*(const bf16x8*)p, *(const bf16x8*)(p + 16));
}

// B fragment (32x16 bf16), tT = B transposed [col][k]: lane l<16: col l K=0..7 / 16..23; +16: K=8..15 / 24..31
__device__ __forceinline__ v16bf b_frag(const bf16* tT, int pitch, int kc) {
    const int lane = threadIdx.x & 31;
    const bf16* p = tT + (size_t)(lane & 15) * pitch + kc + ((lane >> 4) * 8);
    return make_frag(*(const bf16x8*)p, *(const bf16x8*)(p + 16));
}

__device__ __forceinline__ v8f wmma_bf16(v16bf a, v16bf b, v8f c) {
    return __builtin_amdgcn_wmma_f32_16x16x32_bf16(
        false, a, false, b, (short)0, c, false, false);
}

// 32-byte async copy global -> LDS (per lane), tracked by ASYNCcnt.
// INST_OFFSET applies to both LDS and global addresses (ISA 10.7 pseudocode).
__device__ __forceinline__ void async_cp32(unsigned int lds,
                                           unsigned long long gaddr) {
    asm volatile("global_load_async_to_lds_b128 %0, %1, off\n\t"
                 "global_load_async_to_lds_b128 %0, %1, off offset:16"
                 :: "v"(lds), "v"(gaddr) : "memory");
}
__device__ __forceinline__ void wait_async0() {
    asm volatile("s_wait_asynccnt 0x0" ::: "memory");
}

__device__ __forceinline__ float phi_f(float x) {        // elu(x)+1
    return x > 0.0f ? x + 1.0f : __expf(x);
}
__device__ __forceinline__ float gelu_f(float x) {       // tanh approximation
    float x3 = x * x * x;
    return 0.5f * x * (1.0f + tanhf(0.7978845608028654f * (x + 0.044715f * x3)));
}

// ---------------------------------------------------------------------------
// Weight transpose + f32->bf16: WT[n*K + k] = (bf16)W[k*N + n]
// grid (K/8, N/32), block 256
// ---------------------------------------------------------------------------
__global__ __launch_bounds__(256)
void wtrans_kernel(const float* __restrict__ W, bf16* __restrict__ WT,
                   int K, int N) {
    const int k = blockIdx.x * 8 + (threadIdx.x >> 5);
    const int n = blockIdx.y * 32 + (threadIdx.x & 31);
    WT[(size_t)n * K + k] = (bf16)W[(size_t)k * N + n];
}

// Fill vT rows 64..79 per (b,h): row 64 = 1.0 (ksum column), rows 65..79 = 0
__global__ __launch_bounds__(256)
void vext_init_kernel(bf16* __restrict__ vTe) {
    const size_t i = (size_t)blockIdx.x * 256 + threadIdx.x;  // 32*16*NSEQ
    const size_t n  = i & (NSEQ - 1);
    const size_t r  = (i >> 14) & 15;
    const size_t bh = i >> 18;
    vTe[(bh * 80 + 64 + r) * (size_t)NSEQ + n] = (bf16)((r == 0) ? 1.0f : 0.0f);
}

__global__ void zero_kernel(float* __restrict__ p, int n) {
    const int i = blockIdx.x * 256 + threadIdx.x;
    if (i < n) p[i] = 0.0f;
}

// ---------------------------------------------------------------------------
// LayerNorm over D=512: one wave per row; f32 (residual) + bf16 (WMMA) out.
// ---------------------------------------------------------------------------
__global__ __launch_bounds__(256)
void ln_kernel(const float* __restrict__ in, const float* __restrict__ g,
               const float* __restrict__ be, float* __restrict__ out32,
               bf16* __restrict__ outb) {
    const int wave = threadIdx.x >> 5, lane = threadIdx.x & 31;
    const size_t row = (size_t)blockIdx.x * 8 + wave;
    const float* p = in + row * DMODEL;
    float x[16], s = 0.0f, sq = 0.0f;
#pragma unroll
    for (int j = 0; j < 16; ++j) {
        x[j] = p[lane + j * 32];
        s += x[j]; sq += x[j] * x[j];
    }
#pragma unroll
    for (int o = 16; o >= 1; o >>= 1) {
        s  += __shfl_xor(s,  o, 32);
        sq += __shfl_xor(sq, o, 32);
    }
    const float mean = s * (1.0f / DMODEL);
    const float var  = sq * (1.0f / DMODEL) - mean * mean;
    const float rstd = rsqrtf(var + 1e-5f);
#pragma unroll
    for (int j = 0; j < 16; ++j) {
        const int c = lane + j * 32;
        const float v = (x[j] - mean) * rstd * g[c] + be[c];
        out32[row * DMODEL + c] = v;
        outb [row * DMODEL + c] = (bf16)v;
    }
}

// ---------------------------------------------------------------------------
// bf16 WMMA GEMM, 128x128 block tile, 8 waves of 32x64, K-step 32,
// double-buffered LDS filled by async-to-LDS copies.
//  MODE 0: QKV  -> pq row-major (phi), pk^T / v^T transposed packed stores
//  MODE 1: OPROJ-> out32 = res + A@WT + bias        (in-place on res ok)
//  MODE 2: MLP1 -> outb = bf16(gelu(A@WT + bias))
//  MODE 3: MLP2 -> out32 = res + A@WT + bias
// ---------------------------------------------------------------------------
#define BM 128
#define BN 128
#define BK 32

template <int MODE>
__global__ __launch_bounds__(256)
void gemm_kernel(const bf16* __restrict__ A, const bf16* __restrict__ WT,
                 const float* __restrict__ ba, const float* __restrict__ bb,
                 const float* __restrict__ bc,
                 const float* __restrict__ res,
                 float* __restrict__ out32, bf16* __restrict__ outb,
                 bf16* __restrict__ pkT, bf16* __restrict__ vTe,
                 int K, int N) {
    __shared__ __align__(16) bf16 sA [2][BM * BK];   // [m][k]
    __shared__ __align__(16) bf16 sBT[2][BN * BK];   // [n][k]

    const int tid  = threadIdx.x;
    const int lane = tid & 31;
    const int wave = tid >> 5;
    const int wm   = wave >> 1;      // 0..3 : 32 rows each
    const int wn   = wave & 1;       // 0..1 : 64 cols each
    const int mBase = blockIdx.y * BM;
    const int nBase = blockIdx.x * BN;

    // per-thread 32B staging slot (256 threads cover 128x32 halves)
    const bf16* srcA = A  + (size_t)(mBase + (tid >> 1)) * K + (tid & 1) * 16;
    const bf16* srcW = WT + (size_t)(nBase + (tid >> 1)) * K + (tid & 1) * 16;
    const int slot = (tid >> 1) * BK + (tid & 1) * 16;
    const unsigned int ldsA0 = (unsigned int)(uintptr_t)&sA [0][slot];
    const unsigned int ldsA1 = (unsigned int)(uintptr_t)&sA [1][slot];
    const unsigned int ldsW0 = (unsigned int)(uintptr_t)&sBT[0][slot];
    const unsigned int ldsW1 = (unsigned int)(uintptr_t)&sBT[1][slot];

    v8f acc[2][4] = {};

    // prologue: tile 0 into buffer 0
    async_cp32(ldsA0, (unsigned long long)(uintptr_t)srcA);
    async_cp32(ldsW0, (unsigned long long)(uintptr_t)srcW);
    wait_async0();
    __syncthreads();

    int buf = 0;
    for (int kb = 0; kb < K; kb += BK) {
        const int more = (kb + BK < K);
        if (more) {   // overlap next tile copy with this tile's WMMAs
            async_cp32(buf ? ldsA0 : ldsA1,
                       (unsigned long long)(uintptr_t)(srcA + kb + BK));
            async_cp32(buf ? ldsW0 : ldsW1,
                       (unsigned long long)(uintptr_t)(srcW + kb + BK));
        }

        const bf16* aT = sA [buf] + (wm * 32) * BK;
        const bf16* bT = sBT[buf] + (wn * 64) * BK;
        v16bf a0 = a_frag(aT,           BK, 0);
        v16bf a1 = a_frag(aT + 16 * BK, BK, 0);
        v16bf bf0 = b_frag(bT,           BK, 0);
        v16bf bf1 = b_frag(bT + 16 * BK, BK, 0);
        v16bf bf2 = b_frag(bT + 32 * BK, BK, 0);
        v16bf bf3 = b_frag(bT + 48 * BK, BK, 0);
        acc[0][0] = wmma_bf16(a0, bf0, acc[0][0]);
        acc[0][1] = wmma_bf16(a0, bf1, acc[0][1]);
        acc[0][2] = wmma_bf16(a0, bf2, acc[0][2]);
        acc[0][3] = wmma_bf16(a0, bf3, acc[0][3]);
        acc[1][0] = wmma_bf16(a1, bf0, acc[1][0]);
        acc[1][1] = wmma_bf16(a1, bf1, acc[1][1]);
        acc[1][2] = wmma_bf16(a1, bf2, acc[1][2]);
        acc[1][3] = wmma_bf16(a1, bf3, acc[1][3]);

        if (more) {
            wait_async0();        // own copies landed
            __syncthreads();      // everyone's copies visible / reads done
        }
        buf ^= 1;
    }

    // epilogue: lane col = l&15; rows = (l>>4)*8 + r (consecutive tokens)
#pragma unroll
    for (int ms = 0; ms < 2; ++ms)
#pragma unroll
    for (int ns = 0; ns < 4; ++ns) {
        const int col  = nBase + wn * 64 + ns * 16 + (lane & 15);
        const int row0 = mBase + wm * 32 + ms * 16 + (lane >> 4) * 8;
        v8f c = acc[ms][ns];
        if (MODE == 0) {
            const float* bp = col < 512 ? ba : (col < 1024 ? bb : bc);
            const float bvv = bp[col & 511];
            if (col < 512) {                      // pq row-major, phi
#pragma unroll
                for (int r = 0; r < 8; ++r)
                    outb[(size_t)(row0 + r) * 512 + col] =
                        (bf16)phi_f(c[r] + bvv);
            } else {                              // pk^T / v^T packed stores
                const int c2 = col & 511, h = c2 >> 6, d = c2 & 63;
                const size_t bh = (size_t)(row0 >> 14) * 8 + h;
                const size_t n0 = (size_t)(row0 & (NSEQ - 1));
                bf16x8 pk;
                if (col < 1024) {
#pragma unroll
                    for (int r = 0; r < 8; ++r)
                        pk[r] = (bf16)phi_f(c[r] + bvv);
                    *(bf16x8*)(pkT + (bh * 64 + d) * (size_t)NSEQ + n0) = pk;
                } else {
#pragma unroll
                    for (int r = 0; r < 8; ++r)
                        pk[r] = (bf16)(c[r] + bvv);
                    *(bf16x8*)(vTe + (bh * 80 + d) * (size_t)NSEQ + n0) = pk;
                }
            }
        } else if (MODE == 2) {
            const float bvv = ba[col];
#pragma unroll
            for (int r = 0; r < 8; ++r)
                outb[(size_t)(row0 + r) * N + col] = (bf16)gelu_f(c[r] + bvv);
        } else {                                  // MODE 1 / 3
            const float bvv = ba[col];
#pragma unroll
            for (int r = 0; r < 8; ++r) {
                const size_t idx = (size_t)(row0 + r) * N + col;
                out32[idx] = res[idx] + c[r] + bvv;
            }
        }
    }
}

// ---------------------------------------------------------------------------
// kv reduction, LDS-free: per (b,h): kv[d][c] = sum_n pkT[d][n] * vTe[c][n],
// c==64 column of vTe is ones -> ksum. Split over N, f32 global atomics.
// Block = 4 waves; wave owns a 16-row d-tile. Fragments straight from global.
// ---------------------------------------------------------------------------
__global__ __launch_bounds__(128)
void kv_kernel(const bf16* __restrict__ pkT, const bf16* __restrict__ vTe,
               float* __restrict__ kv, int perSplit) {
    const int lane = threadIdx.x & 31, wave = threadIdx.x >> 5;
    const int bh = blockIdx.y;
    const int nStart = blockIdx.x * perSplit;

    const bf16* aBase = pkT + ((size_t)bh * 64 + wave * 16) * NSEQ + nStart;
    const bf16* bBase = vTe + (size_t)bh * 80 * NSEQ + nStart;

    v8f acc[5] = {};
    for (int kb = 0; kb < perSplit; kb += 32) {
        v16bf a = a_frag(aBase + kb, NSEQ, 0);
#pragma unroll
        for (int t = 0; t < 5; ++t) {
            v16bf bf = b_frag(bBase + (size_t)t * 16 * NSEQ + kb, NSEQ, 0);
            acc[t] = wmma_bf16(a, bf, acc[t]);
        }
    }

    float* dst = kv + (size_t)bh * 64 * 80;
    const int d = wave * 16 + (lane >> 4) * 8;
#pragma unroll
    for (int t = 0; t < 5; ++t) {
        const int c = t * 16 + (lane & 15);
#pragma unroll
        for (int r = 0; r < 8; ++r)
            atomicAdd(&dst[(d + r) * 80 + c], acc[t][r]);
    }
}

// ---------------------------------------------------------------------------
// attn = (pq @ kvext); column 64 = denominator -> fused z = 1/(den+eps).
// Block: 8 waves x 32 rows; A frags straight from global pq.
// ---------------------------------------------------------------------------
__global__ __launch_bounds__(256)
void attn_kernel(const bf16* __restrict__ pqb, const float* __restrict__ kv,
                 bf16* __restrict__ attnb) {
    __shared__ __align__(16) bf16 sKvT[80 * 64];   // [c][d]
    const int tid = threadIdx.x, lane = tid & 31, wave = tid >> 5;
    const int bh = blockIdx.y, b = bh >> 3, h = bh & 7;

    const float* src = kv + (size_t)bh * 5120;
    for (int i = tid; i < 5120; i += 256) {
        const int d = i / 80, c = i % 80;
        sKvT[c * 64 + d] = (bf16)src[i];
    }
    __syncthreads();

    const size_t mBase = (size_t)b * NSEQ + (size_t)blockIdx.x * 256 + wave * 32;
    const bf16* pqBase = pqb + mBase * 512 + h * 64;

    v8f acc[2][5] = {};
#pragma unroll
    for (int ks = 0; ks < 2; ++ks) {
        v16bf a0 = a_frag(pqBase,            512, ks * 32);
        v16bf a1 = a_frag(pqBase + 16 * 512, 512, ks * 32);
#pragma unroll
        for (int t = 0; t < 5; ++t) {
            v16bf bf = b_frag(sKvT + t * 16 * 64, 64, ks * 32);
            acc[0][t] = wmma_bf16(a0, bf, acc[0][t]);
            acc[1][t] = wmma_bf16(a1, bf, acc[1][t]);
        }
    }

#pragma unroll
    for (int ms = 0; ms < 2; ++ms) {
        float den[8];
#pragma unroll
        for (int r = 0; r < 8; ++r)   // denominator: tile 4, col 0 of each half
            den[r] = __shfl(acc[ms][4][r], lane & 16, 32);
        const size_t row0 = mBase + ms * 16 + (lane >> 4) * 8;
#pragma unroll
        for (int t = 0; t < 4; ++t) {
            const int c = t * 16 + (lane & 15);
#pragma unroll
            for (int r = 0; r < 8; ++r) {
                const float z = 1.0f / (den[r] + 1e-6f);
                attnb[(row0 + r) * DMODEL + h * 64 + c] =
                    (bf16)(acc[ms][t][r] * z);
            }
        }
    }
}

// ---------------------------------------------------------------------------
extern "C" void kernel_launch(void* const* d_in, const int* in_sizes, int n_in,
                              void* d_out, int out_size, void* d_ws, size_t ws_size,
                              hipStream_t stream) {
    const float* x     = (const float*)d_in[0];
    const float* ln1_g = (const float*)d_in[1];
    const float* ln1_b = (const float*)d_in[2];
    const float* Wq    = (const float*)d_in[3];
    const float* bq    = (const float*)d_in[4];
    const float* Wk    = (const float*)d_in[5];
    const float* bk    = (const float*)d_in[6];
    const float* Wv    = (const float*)d_in[7];
    const float* bv    = (const float*)d_in[8];
    const float* Wo    = (const float*)d_in[9];
    const float* bo    = (const float*)d_in[10];
    const float* ln2_g = (const float*)d_in[11];
    const float* ln2_b = (const float*)d_in[12];
    const float* W1    = (const float*)d_in[13];
    const float* b1    = (const float*)d_in[14];
    const float* W2    = (const float*)d_in[15];
    const float* b2    = (const float*)d_in[16];
    float* out = (float*)d_out;

    // workspace layout (~472 MB)
    char* ws = (char*)d_ws;
    const size_t MB = 1024ull * 1024ull;
    float* h32   = (float*)ws;                    // 128MB: h -> hres -> h2 (in place)
    bf16*  hb    = (bf16*) (ws + 128 * MB);       // 64MB:  ln bf16 (reused for h2)
    bf16*  pqb   = (bf16*) (ws + 192 * MB);       // 64MB:  phi(q) row-major
    bf16*  attnb = (bf16*) (ws + 256 * MB);       // 64MB
    bf16*  pkT   = (bf16*) (ws + 320 * MB);       // 64MB:  [bh][64][NSEQ]
    bf16*  vTe   = (bf16*) (ws + 384 * MB);       // 80MB:  [bh][80][NSEQ]
    bf16*  midb  = (bf16*) (ws + 192 * MB);       // 256MB reuse (after attention)
    float* kvbuf = (float*)(ws + 464 * MB);       // 32*64*80 f32
    bf16*  wqkvT = (bf16*) (ws + 465 * MB);       // [1536][512]
    bf16*  woT   = (bf16*) (ws + 467 * MB);       // [512][512]
    bf16*  w1T   = (bf16*) (ws + 468 * MB);       // [2048][512]
    bf16*  w2T   = (bf16*) (ws + 470 * MB);       // [512][2048]

    // 0) weight transpose + bf16 convert (one pass, ~3.1M elems)
    wtrans_kernel<<<dim3(64,  16), 256, 0, stream>>>(Wq, wqkvT,              512, 512);
    wtrans_kernel<<<dim3(64,  16), 256, 0, stream>>>(Wk, wqkvT + 512 * 512,  512, 512);
    wtrans_kernel<<<dim3(64,  16), 256, 0, stream>>>(Wv, wqkvT + 1024 * 512, 512, 512);
    wtrans_kernel<<<dim3(64,  16), 256, 0, stream>>>(Wo, woT,                512, 512);
    wtrans_kernel<<<dim3(64,  64), 256, 0, stream>>>(W1, w1T,                512, 2048);
    wtrans_kernel<<<dim3(256, 16), 256, 0, stream>>>(W2, w2T,               2048, 512);
    vext_init_kernel<<<(32 * 16 * NSEQ) / 256, 256, 0, stream>>>(vTe);

    // 1) LN1
    ln_kernel<<<MTOK / 8, 256, 0, stream>>>(x, ln1_g, ln1_b, h32, hb);

    // 2) QKV projection: pq (phi, row-major) + pk^T/v^T (transposed)
    gemm_kernel<0><<<dim3(1536 / BN, MTOK / BM), 256, 0, stream>>>(
        hb, wqkvT, bq, bk, bv, nullptr, nullptr, pqb, pkT, vTe, 512, 1536);

    // 3) kv / ksum reduction
    zero_kernel<<<(32 * 64 * 80 + 255) / 256, 256, 0, stream>>>(kvbuf, 32 * 64 * 80);
    kv_kernel<<<dim3(16, 32), 128, 0, stream>>>(pkT, vTe, kvbuf, NSEQ / 16);

    // 4) attn = (pq @ kvext) * 1/(den+eps)
    attn_kernel<<<dim3(NSEQ / 256, 32), 256, 0, stream>>>(pqb, kvbuf, attnb);

    // 5) O-projection + residual (in place on h32)
    gemm_kernel<1><<<dim3(DMODEL / BN, MTOK / BM), 256, 0, stream>>>(
        attnb, woT, bo, nullptr, nullptr, h32, h32, nullptr, nullptr, nullptr,
        512, DMODEL);

    // 6) LN2 (in place f32, bf16 into hb)
    ln_kernel<<<MTOK / 8, 256, 0, stream>>>(h32, ln2_g, ln2_b, h32, hb);

    // 7) MLP1 + gelu
    gemm_kernel<2><<<dim3(HID / BN, MTOK / BM), 256, 0, stream>>>(
        hb, w1T, b1, nullptr, nullptr, nullptr, nullptr, midb, nullptr, nullptr,
        512, HID);

    // 8) MLP2 + residual -> d_out
    gemm_kernel<3><<<dim3(DMODEL / BN, MTOK / BM), 256, 0, stream>>>(
        midb, w2T, b2, nullptr, nullptr, h32, out, nullptr, nullptr, nullptr,
        HID, DMODEL);
}
